// MultiHeadSelfAttention_8581344657854
// MI455X (gfx1250) — compile-verified, matrix operand order checked by reading, not run on hardware
//
#include <hip/hip_runtime.h>
#include <hip/hip_bf16.h>

// ---------------------------------------------------------------------------
// MI455X (gfx1250) fused causal MHA, bf16 WMMA path + TDM-staged flash attn.
//   - All matmuls: v_wmma_f32_16x16x32_bf16 (fp32 accumulate).
//   - Flash attention: 8 waves/block share one (b,h,128-query) tile; K/V
//     64-key chunks are DMA'd into double-buffered LDS by the Tensor Data
//     Mover (tensor_load_to_lds), synchronized with s_wait_tensorcnt +
//     one workgroup barrier per chunk. ~8x L2-traffic reduction vs per-wave
//     loads, DMA overlapped with WMMA.
// Workspace ~88MB: xb 16 | weights 8 | Q,K 32 | Vt 16 | AO 16.
// ---------------------------------------------------------------------------

typedef __attribute__((ext_vector_type(16))) __bf16 v16bf;
typedef __attribute__((ext_vector_type(8)))  __bf16 v8bf;
typedef __attribute__((ext_vector_type(8)))  float  v8f;
typedef unsigned int u32x4 __attribute__((ext_vector_type(4)));
typedef int          i32x4 __attribute__((ext_vector_type(4)));
typedef int          i32x8 __attribute__((ext_vector_type(8)));

#define BATCH   4
#define HEADS   16
#define SEQ     2048
#define DMODEL  1024
#define DK      64
#define ROWS    (BATCH*SEQ)

// Build a v16bf WMMA operand from two contiguous 16-byte chunks.
// Matches the 16-bit 16x32 A layout (and symmetric B layout):
// elements 0..7  = K = base + hi*8 .. +7 ; elements 8..15 = K = base+16+hi*8 ..
static __device__ inline v16bf load2(const __bf16* p0, const __bf16* p1) {
  union { v16bf v; v8bf h[2]; } u;
  u.h[0] = *(const v8bf*)p0;
  u.h[1] = *(const v8bf*)p1;
  return u.v;
}

// ---- Tensor Data Mover descriptor helpers (CDNA5 ISA)
// Group0: count=1 | lds_addr | global_addr[56:0] | type=2 ("image")
static __device__ inline u32x4 tdm_g0(unsigned int lds, unsigned long long ga) {
  u32x4 g;
  g.x = 1u;                                   // count=1, user descriptor
  g.y = lds;                                  // LDS byte address
  g.z = (unsigned int)ga;                     // global_addr[31:0]
  g.w = (unsigned int)(ga >> 32) | (2u << 30);// global_addr[56:32] | type=2
  return g;
}
// Group1: 2D tile, data_size=2B, no multicast/pad/iterate.
static __device__ inline i32x8 tdm_g1(unsigned int dim0, unsigned int dim1,
                                      unsigned int tile0, unsigned int tile1,
                                      unsigned long long stride0) {
  i32x8 g = {};
  g[0] = 0x00010000;                                  // data_size=1 (2 bytes)
  g[1] = (int)((dim0 & 0xffffu) << 16);               // tensor_dim0[15:0]
  g[2] = (int)((dim0 >> 16) | ((dim1 & 0xffffu) << 16));
  g[3] = (int)((dim1 >> 16) | (tile0 << 16));         // tile_dim0
  g[4] = (int)tile1;                                  // tile_dim1 (tile_dim2=0)
  g[5] = (int)(stride0 & 0xffffffffull);              // tensor_dim0_stride lo
  g[6] = (int)(stride0 >> 32);                        // stride hi (dim1_stride=0)
  g[7] = 0;
  return g;
}
static __device__ inline void tdm_load(u32x4 g0, i32x8 g1) {
  i32x4 z4 = {};
#if defined(__clang_major__) && __clang_major__ >= 23
  i32x8 z8 = {};
  __builtin_amdgcn_tensor_load_to_lds(g0, g1, z4, z4, z8, 0);
#else
  __builtin_amdgcn_tensor_load_to_lds(g0, g1, z4, z4, 0);
#endif
}

__global__ void cvt_bf16(const float* __restrict__ src, __bf16* __restrict__ dst, int n) {
  int i = blockIdx.x * blockDim.x + threadIdx.x;
  if (i < n) dst[i] = (__bf16)src[i];
}

// ---------------------------------------------------------------------------
// GEMM: Y[r,e] = sum_d A[r,d] * W[e,d]  (bf16 in, fp32 acc). Wave tile 32x64.
// mode 0/1: Q/K -> bf16 [b,h,s,d]; mode 2: Vt -> bf16 [b,h,d,s] (16B packed);
// mode 3: fp32 [b,s,e] final output.
// ---------------------------------------------------------------------------
__global__ void __launch_bounds__(256)
gemm_proj(const __bf16* __restrict__ A, const __bf16* __restrict__ W,
          __bf16* __restrict__ outQK, __bf16* __restrict__ outVt,
          float* __restrict__ outF, int mode)
{
  const int wid      = (blockIdx.x * 256 + (int)threadIdx.x) >> 5;
  const int lane     = threadIdx.x & 31;
  const int rowTile  = wid >> 4;
  const int colChunk = wid & 15;
  const int row0     = rowTile * 32;
  const int e0       = colChunk * 64;
  const int n        = lane & 15;
  const int hi       = lane >> 4;

  const __bf16* a0p = A + (size_t)(row0 + n) * DMODEL;
  const __bf16* a1p = a0p + (size_t)16 * DMODEL;
  const __bf16* wp[4];
#pragma unroll
  for (int c = 0; c < 4; ++c) wp[c] = W + (size_t)(e0 + c * 16 + n) * DMODEL;

  v8f zero = {};
  v8f acc[2][4];
#pragma unroll
  for (int r = 0; r < 2; ++r)
#pragma unroll
    for (int c = 0; c < 4; ++c) acc[r][c] = zero;

  for (int k = 0; k < DMODEL; k += 32) {
    const int o0 = k + hi * 8;
    const int o1 = k + 16 + hi * 8;
    v16bf a0 = load2(a0p + o0, a0p + o1);
    v16bf a1 = load2(a1p + o0, a1p + o1);
#pragma unroll
    for (int c = 0; c < 4; ++c) {
      v16bf b = load2(wp[c] + o0, wp[c] + o1);
      acc[0][c] = __builtin_amdgcn_wmma_f32_16x16x32_bf16(false, a0, false, b,
                                                          (short)0, acc[0][c], false, false);
      acc[1][c] = __builtin_amdgcn_wmma_f32_16x16x32_bf16(false, a1, false, b,
                                                          (short)0, acc[1][c], false, false);
    }
  }

  const int b     = row0 >> 11;
  const int sbase = row0 & (SEQ - 1);

  if (mode <= 1) {
    __bf16* base = outQK + ((size_t)(b * HEADS + colChunk)) * SEQ * DK;
#pragma unroll
    for (int r = 0; r < 2; ++r)
#pragma unroll
      for (int c = 0; c < 4; ++c) {
        const int d = c * 16 + n;
#pragma unroll
        for (int j = 0; j < 8; ++j) {
          const int s = sbase + r * 16 + hi * 8 + j;
          base[(size_t)s * DK + d] = (__bf16)acc[r][c][j];
        }
      }
  } else if (mode == 2) {
#pragma unroll
    for (int r = 0; r < 2; ++r)
#pragma unroll
      for (int c = 0; c < 4; ++c) {
        const int d  = c * 16 + n;
        const int s0 = sbase + r * 16 + hi * 8;
        v8bf pk;
#pragma unroll
        for (int j = 0; j < 8; ++j) pk[j] = (__bf16)acc[r][c][j];
        *(v8bf*)(outVt + ((size_t)(b * HEADS + colChunk) * DK + d) * SEQ + s0) = pk;
      }
  } else {
#pragma unroll
    for (int r = 0; r < 2; ++r)
#pragma unroll
      for (int c = 0; c < 4; ++c) {
        const int e = e0 + c * 16 + n;
#pragma unroll
        for (int j = 0; j < 8; ++j) {
          const int rr = row0 + r * 16 + hi * 8 + j;
          outF[(size_t)rr * DMODEL + e] = acc[r][c][j];
        }
      }
  }
}

// ---------------------------------------------------------------------------
// RoPE in-place on Q and K: [B,H,S,64] bf16, one thread per (b,h,s,pair).
// ---------------------------------------------------------------------------
__global__ void rope_inplace(__bf16* __restrict__ Q, __bf16* __restrict__ K) {
  const int i = blockIdx.x * blockDim.x + threadIdx.x;
  const int p = i & 31;
  const int s = (i >> 5) & (SEQ - 1);
  const int h = (i >> 16) & (HEADS - 1);
  const int b = i >> 20;
  const float inv = exp2f((float)(-2 * p) * (13.287712379549449f / 64.0f));
  const float ang = (float)s * inv;
  float sn, cs;
  __sincosf(ang, &sn, &cs);
  const size_t base = (((size_t)(b * HEADS + h)) * SEQ + s) * DK + 2 * p;
  float q1 = (float)Q[base], q2 = (float)Q[base + 1];
  Q[base]     = (__bf16)(q1 * cs - q2 * sn);
  Q[base + 1] = (__bf16)(q1 * sn + q2 * cs);
  float k1 = (float)K[base], k2 = (float)K[base + 1];
  K[base]     = (__bf16)(k1 * cs - k2 * sn);
  K[base + 1] = (__bf16)(k1 * sn + k2 * cs);
}

// ---------------------------------------------------------------------------
// Block-cooperative flash attention with TDM-staged K/V.
// Block = 8 waves = one (b, h, 128-query) tile; wave wib owns queries
// [q0, q0+16). Key loop in 64-key chunks, double-buffered in LDS:
//   iter ch: wave0 s_wait_tensorcnt(0)  -> buf[cur] DMA complete
//            __syncthreads()            -> publish buf[cur]; all done w/ prev
//            wave0 issues TDM for ch+1 -> buf[cur^1]
//            all waves: WMMA scores + online softmax + WMMA PV on buf[cur]
// ---------------------------------------------------------------------------
__global__ void __launch_bounds__(256)
flash_attn(const __bf16* __restrict__ Q, const __bf16* __restrict__ K,
           const __bf16* __restrict__ Vt, __bf16* __restrict__ AO)
{
  __shared__ __align__(16) __bf16 ldsK[2][64 * 64];   // [key][d]   8KB x2
  __shared__ __align__(16) __bf16 ldsV[2][64 * 64];   // [d][key]   8KB x2
  __shared__ __align__(16) __bf16 ptile[8][16 * 72];  // per-wave P, stride 72

  const int wib  = threadIdx.x >> 5;
  const int lane = threadIdx.x & 31;
  const int qb   = blockIdx.x & 15;          // 128-query block within sequence
  const int h    = (blockIdx.x >> 4) & 15;
  const int b    = blockIdx.x >> 8;
  const int q0   = qb * 128 + wib * 16;
  const int n    = lane & 15;
  const int hi   = lane >> 4;
  __bf16* myp = &ptile[wib][0];

  const __bf16* Kbase = K  + ((size_t)(b * HEADS + h)) * SEQ * DK;
  const __bf16* Vbase = Vt + ((size_t)(b * HEADS + h)) * DK * SEQ;

  const __bf16* Qr = Q + (((size_t)(b * HEADS + h)) * SEQ + q0 + n) * DK;
  v16bf aq0 = load2(Qr + hi * 8,      Qr + 16 + hi * 8);
  v16bf aq1 = load2(Qr + 32 + hi * 8, Qr + 48 + hi * 8);

  v8f zero = {};
  v8f acc[4];
  float mrun[8], lrun[8];
#pragma unroll
  for (int c = 0; c < 4; ++c) acc[c] = zero;
#pragma unroll
  for (int j = 0; j < 8; ++j) { mrun[j] = -3.0e38f; lrun[j] = 0.0f; }

  const int nch = 2 * qb + 2;                 // 64-key chunks covering q0b+127

  // Prologue: DMA chunk 0 into buffer 0 (one wave issues; EXEC ignored by TDM).
  if (wib == 0) {
    tdm_load(tdm_g0((unsigned int)(unsigned long long)(uintptr_t)&ldsK[0][0],
                    (unsigned long long)(uintptr_t)Kbase),
             tdm_g1(DK, 1u << 20, DK, 64, DK));
    tdm_load(tdm_g0((unsigned int)(unsigned long long)(uintptr_t)&ldsV[0][0],
                    (unsigned long long)(uintptr_t)Vbase),
             tdm_g1(SEQ, DK, 64, DK, SEQ));
  }

  for (int ch = 0; ch < nch; ++ch) {
    const int cur = ch & 1;
    const int kc  = ch * 64;

    if (wib == 0) __builtin_amdgcn_s_wait_tensorcnt(0);
    __syncthreads();                       // buf[cur] ready; buf[cur^1] free

    if (wib == 0 && ch + 1 < nch) {
      const int kn = (ch + 1) * 64;
      tdm_load(tdm_g0((unsigned int)(unsigned long long)(uintptr_t)&ldsK[cur ^ 1][0],
                      (unsigned long long)(uintptr_t)(Kbase + (size_t)kn * DK)),
               tdm_g1(DK, 1u << 20, DK, 64, DK));
      tdm_load(tdm_g0((unsigned int)(unsigned long long)(uintptr_t)&ldsV[cur ^ 1][0],
                      (unsigned long long)(uintptr_t)(Vbase + kn)),
               tdm_g1(SEQ, DK, 64, DK, SEQ));
    }

    if (kc <= q0 + 15) {                   // this wave still has live keys here
      const __bf16* lk = &ldsK[cur][0];
      const __bf16* lv = &ldsV[cur][0];

      // ---- scores: 4 subtiles of 16 keys --------------------------------
      v8f sv[4];
#pragma unroll
      for (int t = 0; t < 4; ++t) {
        const int k0 = kc + t * 16;
        if (k0 <= q0 + 15) {
          const __bf16* Kr = lk + (size_t)(t * 16 + n) * DK;
          v16bf bk0 = load2(Kr + hi * 8,      Kr + 16 + hi * 8);
          v16bf bk1 = load2(Kr + 32 + hi * 8, Kr + 48 + hi * 8);
          v8f s = __builtin_amdgcn_wmma_f32_16x16x32_bf16(false, aq0, false, bk0,
                                                          (short)0, zero, false, false);
          s = __builtin_amdgcn_wmma_f32_16x16x32_bf16(false, aq1, false, bk1,
                                                      (short)0, s, false, false);
#pragma unroll
          for (int j = 0; j < 8; ++j) {
            const int qg = q0 + j + hi * 8;
            const int kg = k0 + n;
            s[j] = (kg <= qg) ? s[j] * 0.125f : -1.0e30f;
          }
          sv[t] = s;
        } else {
#pragma unroll
          for (int j = 0; j < 8; ++j) sv[t][j] = -1.0e30f;
        }
      }

      // ---- online softmax over 64 keys ----------------------------------
#pragma unroll
      for (int j = 0; j < 8; ++j) {
        float v = fmaxf(fmaxf(sv[0][j], sv[1][j]), fmaxf(sv[2][j], sv[3][j]));
#pragma unroll
        for (int off = 1; off < 16; off <<= 1) v = fmaxf(v, __shfl_xor(v, off, 32));
        const float mnew = fmaxf(mrun[j], v);
        const float corr = __expf(mrun[j] - mnew);
        mrun[j] = mnew;
        float pv0 = __expf(sv[0][j] - mnew);
        float pv1 = __expf(sv[1][j] - mnew);
        float pv2 = __expf(sv[2][j] - mnew);
        float pv3 = __expf(sv[3][j] - mnew);
        float rs = (pv0 + pv1) + (pv2 + pv3);
#pragma unroll
        for (int off = 1; off < 16; off <<= 1) rs += __shfl_xor(rs, off, 32);
        lrun[j] = lrun[j] * corr + rs;
#pragma unroll
        for (int c = 0; c < 4; ++c) acc[c][j] *= corr;
        const int prow = (j + hi * 8) * 72;
        myp[prow + n]      = (__bf16)pv0;
        myp[prow + 16 + n] = (__bf16)pv1;
        myp[prow + 32 + n] = (__bf16)pv2;
        myp[prow + 48 + n] = (__bf16)pv3;
      }

      asm volatile("s_wait_dscnt 0" ::: "memory");   // same-wave DS in-order

      // ---- P(16x64) x V(64x64) ------------------------------------------
      v16bf ap0 = load2(myp + (size_t)n * 72 + hi * 8,
                        myp + (size_t)n * 72 + 16 + hi * 8);
      v16bf ap1 = load2(myp + (size_t)n * 72 + 32 + hi * 8,
                        myp + (size_t)n * 72 + 48 + hi * 8);
#pragma unroll
      for (int c = 0; c < 4; ++c) {
        const __bf16* Vr = lv + (size_t)(c * 16 + n) * 64;
        v16bf bv0 = load2(Vr + hi * 8,      Vr + 16 + hi * 8);
        v16bf bv1 = load2(Vr + 32 + hi * 8, Vr + 48 + hi * 8);
        acc[c] = __builtin_amdgcn_wmma_f32_16x16x32_bf16(false, ap0, false, bv0,
                                                         (short)0, acc[c], false, false);
        acc[c] = __builtin_amdgcn_wmma_f32_16x16x32_bf16(false, ap1, false, bv1,
                                                         (short)0, acc[c], false, false);
      }
    }
  }

  // Normalize and store attention output bf16 [b, s, h*64+d].
#pragma unroll
  for (int j = 0; j < 8; ++j) {
    const float rinv = 1.0f / lrun[j];
    const int s = q0 + j + hi * 8;
#pragma unroll
    for (int c = 0; c < 4; ++c) {
      const int e = h * DK + c * 16 + n;
      AO[((size_t)(b * SEQ + s)) * DMODEL + e] = (__bf16)(acc[c][j] * rinv);
    }
  }
}

// ---------------------------------------------------------------------------
extern "C" void kernel_launch(void* const* d_in, const int* in_sizes, int n_in,
                              void* d_out, int out_size, void* d_ws, size_t ws_size,
                              hipStream_t stream)
{
  const float* x  = (const float*)d_in[0];
  const float* PQ = (const float*)d_in[1];
  const float* PK = (const float*)d_in[2];
  const float* PV = (const float*)d_in[3];
  const float* PO = (const float*)d_in[4];

  char* ws = (char*)d_ws;
  size_t off = 0;
  __bf16* xb  = (__bf16*)(ws + off); off += (size_t)ROWS * DMODEL * 2;
  __bf16* pqb = (__bf16*)(ws + off); off += (size_t)DMODEL * DMODEL * 2;
  __bf16* pkb = (__bf16*)(ws + off); off += (size_t)DMODEL * DMODEL * 2;
  __bf16* pvb = (__bf16*)(ws + off); off += (size_t)DMODEL * DMODEL * 2;
  __bf16* pob = (__bf16*)(ws + off); off += (size_t)DMODEL * DMODEL * 2;
  __bf16* Qb  = (__bf16*)(ws + off); off += (size_t)BATCH * HEADS * SEQ * DK * 2;
  __bf16* Kb  = (__bf16*)(ws + off); off += (size_t)BATCH * HEADS * SEQ * DK * 2;
  __bf16* Vtb = (__bf16*)(ws + off); off += (size_t)BATCH * HEADS * DK * SEQ * 2;
  __bf16* AOb = (__bf16*)(ws + off); off += (size_t)ROWS * DMODEL * 2;

  const int nx = ROWS * DMODEL;
  const int nw = DMODEL * DMODEL;
  cvt_bf16<<<nx / 256, 256, 0, stream>>>(x,  xb,  nx);
  cvt_bf16<<<nw / 256, 256, 0, stream>>>(PQ, pqb, nw);
  cvt_bf16<<<nw / 256, 256, 0, stream>>>(PK, pkb, nw);
  cvt_bf16<<<nw / 256, 256, 0, stream>>>(PV, pvb, nw);
  cvt_bf16<<<nw / 256, 256, 0, stream>>>(PO, pob, nw);

  gemm_proj<<<512, 256, 0, stream>>>(xb, pqb, Qb, nullptr, nullptr, 0);
  gemm_proj<<<512, 256, 0, stream>>>(xb, pkb, Kb, nullptr, nullptr, 1);
  gemm_proj<<<512, 256, 0, stream>>>(xb, pvb, nullptr, Vtb, nullptr, 2);

  rope_inplace<<<(BATCH * HEADS * SEQ * 32) / 256, 256, 0, stream>>>(Qb, Kb);

  // B*H*(S/128) = 1024 blocks of 8 waves
  flash_attn<<<BATCH * HEADS * (SEQ / 128), 256, 0, stream>>>(Qb, Kb, Vtb, AOb);

  gemm_proj<<<512, 256, 0, stream>>>(AOb, pob, nullptr, nullptr, (float*)d_out, 3);
}
